// RNNModel_64476049047572
// MI455X (gfx1250) — compile-verified, matrix operand order checked by reading, not run on hardware
//
#include <hip/hip_runtime.h>

// ---------------------------------------------------------------------------
// 2-layer LSTM (B=64, T=256, IN=H=OUT=1024) + final FC, for gfx1250 (MI455X).
//   * bf16 WMMA (v_wmma_f32_16x16x32_bf16), f32 accumulation. 32MB of bf16
//     weights stay resident in the 192MB L2 across the whole recurrence.
//   * One fused kernel per (t, layer): gate GEMM + bias + sigmoid/tanh c/h
//     update in a single launch; 512 dependent launches + FC, graph-captured.
//   * CDNA5 async data movement: the recurrent h panel (64x1024 bf16) is
//     staged into LDS with GLOBAL_LOAD_ASYNC_TO_LDS_B128 at kernel entry,
//     overlapped with the x-half of the K reduction, then consumed from LDS
//     (ds_load_b128) for the h-half. s_wait_asynccnt + barrier synchronize.
// ---------------------------------------------------------------------------

#define BB   64
#define TT   256
#define HH   1024
#define OUTN 1024
#define RS   1032   // padded LDS row stride (elements) to spread banks

typedef __attribute__((ext_vector_type(16))) __bf16 v16bf;
typedef __attribute__((ext_vector_type(8)))  __bf16 bf16x8;
typedef __attribute__((ext_vector_type(8)))  float  v8f;

// WMMA A/B operand for one 32-K slab: lane holds two 8-element (16B) chunks
// at +0 and +16 elements (the +8*sel shift is folded into the row pointer).
__device__ __forceinline__ v16bf load_frag(const __bf16* __restrict__ p) {
    bf16x8 lo = *(const bf16x8*)(p);
    bf16x8 hi = *(const bf16x8*)(p + 16);
    return __builtin_shufflevector(lo, hi, 0, 1, 2, 3, 4, 5, 6, 7,
                                           8, 9, 10, 11, 12, 13, 14, 15);
}

__device__ __forceinline__ float sigf(float x) {
    return 1.0f / (1.0f + __expf(-x));
}

// Async 16-byte copy global -> LDS (ASYNCcnt-tracked). lds_byte is the LDS
// byte address (low 32 bits of the generic shared pointer).
__device__ __forceinline__ void async_copy16(unsigned lds_byte,
                                             const __bf16* gptr) {
    asm volatile("global_load_async_to_lds_b128 %0, %1, off"
                 :: "v"(lds_byte), "v"(gptr)
                 : "memory");
}

// ---------------------------------------------------------------------------
// fp32 -> bf16 conversion (grid-stride)
// ---------------------------------------------------------------------------
__global__ void f32_to_bf16_kernel(const float* __restrict__ src,
                                   __bf16* __restrict__ dst, int n) {
    int i = blockIdx.x * blockDim.x + threadIdx.x;
    int stride = gridDim.x * blockDim.x;
    for (; i < n; i += stride) dst[i] = (__bf16)src[i];
}

// ---------------------------------------------------------------------------
// Fused LSTM step for one layer at one timestep.
// Grid: H/16 = 64 blocks. Block: 512 threads = 16 waves = 4 m-tiles x 4 gates.
// ---------------------------------------------------------------------------
__global__ __launch_bounds__(512)
void lstm_step_kernel(const __bf16* __restrict__ X, int ldx,
                      const __bf16* __restrict__ Wx, int K1,
                      const __bf16* __restrict__ Wh,
                      const float* __restrict__ bias,
                      const __bf16* __restrict__ h_in,
                      __bf16* __restrict__ h_out,
                      float* __restrict__ c) {
    __shared__ __bf16 hp[BB][RS];      // staged h panel: ~129 KB
    __shared__ float gsm[4][BB][16];   // f,i,g,o gate tiles: 16 KB

    const int tid  = threadIdx.x;
    const int wave = tid >> 5;
    const int lane = tid & 31;
    const int l15  = lane & 15;
    const int sel  = lane >> 4;        // which K-half this lane holds
    const int mt   = wave >> 2;        // batch tile 0..3
    const int g    = wave & 3;         // gate 0..3 (f,i,g,o split order)
    const int n0   = blockIdx.x << 4;  // hidden-unit block
    const int m0   = mt << 4;

    // ---- Kick off async staging of the whole h panel (64 x 1024 bf16).
    // 8192 16-byte chunks / 512 threads = 16 async b128 copies per thread.
    {
        const unsigned hb = (unsigned)(size_t)&hp[0][0];
#pragma unroll
        for (int i = 0; i < 16; ++i) {
            const int q   = tid + (i << 9);
            const int row = q >> 7;
            const int col = (q & 127) << 3;
            async_copy16(hb + (unsigned)(row * RS + col) * 2u,
                         h_in + (size_t)row * HH + col);
        }
    }

    const int j = g * HH + n0 + l15;   // gate row in Wx/Wh and bias index

    const __bf16* arow  = X  + (size_t)(m0 + l15) * ldx + sel * 8;
    const __bf16* bxrow = Wx + (size_t)j * K1           + sel * 8;
    const __bf16* bhrow = Wh + (size_t)j * HH           + sel * 8;

    const float bj = bias[j];
    v8f acc;
#pragma unroll
    for (int r = 0; r < 8; ++r) acc[r] = bj;

    // ---- x-half of the reduction (from global), overlapped with staging:
    // gates += xt @ Wx^T
#pragma unroll 4
    for (int k0 = 0; k0 < K1; k0 += 32) {
        v16bf a = load_frag(arow + k0);
        v16bf b = load_frag(bxrow + k0);
        acc = __builtin_amdgcn_wmma_f32_16x16x32_bf16(
            false, a, false, b, (short)0, acc, false, false);
    }

    // ---- Wait for the async h panel, make it visible block-wide.
    asm volatile("s_wait_asynccnt 0x0" ::: "memory");
    __syncthreads();

    // ---- h-half of the reduction, A-fragments from LDS: gates += h @ Wh^T
    const __bf16* hrow = &hp[m0 + l15][sel * 8];
#pragma unroll 4
    for (int k0 = 0; k0 < HH; k0 += 32) {
        v16bf a = load_frag(hrow + k0);
        v16bf b = load_frag(bhrow + k0);
        acc = __builtin_amdgcn_wmma_f32_16x16x32_bf16(
            false, a, false, b, (short)0, acc, false, false);
    }

    // C/D layout: VGPR r -> M = m0 + 8*sel + r, N = l15
#pragma unroll
    for (int r = 0; r < 8; ++r)
        gsm[g][m0 + sel * 8 + r][l15] = acc[r];

    __syncthreads();

    // ---- Pointwise LSTM cell update for the 64x16 slice this block owns.
    for (int e = tid; e < BB * 16; e += 512) {
        const int brw = e >> 4;
        const int n   = e & 15;
        const float f  = gsm[0][brw][n];
        const float ii = gsm[1][brw][n];
        const float gg = gsm[2][brw][n];
        const float o  = gsm[3][brw][n];
        const size_t idx = (size_t)brw * HH + n0 + n;
        const float cold = c[idx];
        const float cn = sigf(f) * cold + sigf(ii) * tanhf(gg);
        c[idx] = cn;
        h_out[idx] = (__bf16)(sigf(o) * tanhf(cn));
    }
}

// ---------------------------------------------------------------------------
// Final FC: out[64 x 1024] = h1 @ Wfc^T + bfc   (f32 output)
// Grid: OUT/16 = 64 blocks x 128 threads (4 waves = 4 batch tiles).
// ---------------------------------------------------------------------------
__global__ __launch_bounds__(128)
void fc_kernel(const __bf16* __restrict__ Hfin,
               const __bf16* __restrict__ Wfc,
               const float* __restrict__ bfc,
               float* __restrict__ out) {
    const int tid  = threadIdx.x;
    const int wave = tid >> 5;         // batch tile
    const int lane = tid & 31;
    const int l15  = lane & 15;
    const int sel  = lane >> 4;
    const int n0   = blockIdx.x << 4;
    const int m0   = wave << 4;

    const __bf16* arow = Hfin + (size_t)(m0 + l15) * HH + sel * 8;
    const __bf16* brow = Wfc  + (size_t)(n0 + l15) * HH + sel * 8;

    const float bj = bfc[n0 + l15];
    v8f acc;
#pragma unroll
    for (int r = 0; r < 8; ++r) acc[r] = bj;

#pragma unroll 4
    for (int k0 = 0; k0 < HH; k0 += 32) {
        v16bf a = load_frag(arow + k0);
        v16bf b = load_frag(brow + k0);
        acc = __builtin_amdgcn_wmma_f32_16x16x32_bf16(
            false, a, false, b, (short)0, acc, false, false);
    }

#pragma unroll
    for (int r = 0; r < 8; ++r)
        out[(size_t)(m0 + sel * 8 + r) * OUTN + n0 + l15] = acc[r];
}

// ---------------------------------------------------------------------------
// Host-side launcher
// ---------------------------------------------------------------------------
extern "C" void kernel_launch(void* const* d_in, const int* in_sizes, int n_in,
                              void* d_out, int out_size, void* d_ws, size_t ws_size,
                              hipStream_t stream) {
    const float* x    = (const float*)d_in[0];   // [B, T, IN]
    const float* Wx0  = (const float*)d_in[1];   // [4H, IN]
    const float* Wh0  = (const float*)d_in[2];   // [4H, H]
    const float* b0   = (const float*)d_in[3];   // [4H]
    const float* Wx1  = (const float*)d_in[4];   // [4H, H]
    const float* Wh1  = (const float*)d_in[5];   // [4H, H]
    const float* b1   = (const float*)d_in[6];   // [4H]
    const float* Wfc  = (const float*)d_in[7];   // [OUT, H]
    const float* bfc  = (const float*)d_in[8];   // [OUT]
    float* out        = (float*)d_out;           // [B, OUT]

    // Workspace carve-up (256B aligned). ~70 MB total.
    char* ws = (char*)d_ws;
    size_t off = 0;
    auto carve = [&](size_t bytes) -> void* {
        void* p = ws + off;
        off += (bytes + 255) & ~(size_t)255;
        return p;
    };
    const size_t nX  = (size_t)BB * TT * HH;      // 16.7M
    const size_t nW  = (size_t)4 * HH * HH;       // 4.19M per weight matrix
    const size_t nWf = (size_t)OUTN * HH;         // 1M
    const size_t nS  = (size_t)BB * HH;           // 64K state elems

    __bf16* Xbf  = (__bf16*)carve(nX * 2);
    __bf16* Wx0b = (__bf16*)carve(nW * 2);
    __bf16* Wh0b = (__bf16*)carve(nW * 2);
    __bf16* Wx1b = (__bf16*)carve(nW * 2);
    __bf16* Wh1b = (__bf16*)carve(nW * 2);
    __bf16* Wfcb = (__bf16*)carve(nWf * 2);
    __bf16* h0a  = (__bf16*)carve(nS * 2);
    __bf16* h0b  = (__bf16*)carve(nS * 2);
    __bf16* h1a  = (__bf16*)carve(nS * 2);
    __bf16* h1b  = (__bf16*)carve(nS * 2);
    float*  c0   = (float*)carve(nS * 4);
    float*  c1   = (float*)carve(nS * 4);

    // bf16 conversions (weights stay L2-resident for the whole recurrence)
    f32_to_bf16_kernel<<<4096, 256, 0, stream>>>(x,   Xbf,  (int)nX);
    f32_to_bf16_kernel<<<2048, 256, 0, stream>>>(Wx0, Wx0b, (int)nW);
    f32_to_bf16_kernel<<<2048, 256, 0, stream>>>(Wh0, Wh0b, (int)nW);
    f32_to_bf16_kernel<<<2048, 256, 0, stream>>>(Wx1, Wx1b, (int)nW);
    f32_to_bf16_kernel<<<2048, 256, 0, stream>>>(Wh1, Wh1b, (int)nW);
    f32_to_bf16_kernel<<<1024, 256, 0, stream>>>(Wfc, Wfcb, (int)nWf);

    // Zero initial state (capturable stream memsets)
    hipMemsetAsync(h0a, 0, nS * 2, stream);
    hipMemsetAsync(h1a, 0, nS * 2, stream);
    hipMemsetAsync(c0,  0, nS * 4, stream);
    hipMemsetAsync(c1,  0, nS * 4, stream);

    __bf16* h0p[2] = {h0a, h0b};
    __bf16* h1p[2] = {h1a, h1b};

    // Sequential recurrence: kernel boundaries provide the grid-wide sync.
    for (int t = 0; t < TT; ++t) {
        const __bf16* xt = Xbf + (size_t)t * HH;   // row stride T*IN
        lstm_step_kernel<<<HH / 16, 512, 0, stream>>>(
            xt, TT * HH, Wx0b, HH, Wh0b, b0,
            h0p[t & 1], h0p[(t + 1) & 1], c0);
        lstm_step_kernel<<<HH / 16, 512, 0, stream>>>(
            h0p[(t + 1) & 1], HH, Wx1b, HH, Wh1b, b1,
            h1p[t & 1], h1p[(t + 1) & 1], c1);
    }

    // Final FC on last timestep's top-layer hidden (lands in h1p[0] after 256 steps)
    fc_kernel<<<OUTN / 16, 128, 0, stream>>>(h1p[0], Wfcb, bfc, out);
}
